// VQEmbedding_71571335020768
// MI455X (gfx1250) — matrix operand order repaired and checked by reading, not compiled, not run on hardware
//
#include <hip/hip_runtime.h>

// ---------------------------------------------------------------------------
// VQ codebook lookup for MI455X (gfx1250, wave32, WMMA).
//   x:        [8, 4096, 512] f32   (BS = 32768 rows)
//   codebook: [2048, 512]    f32
// Outputs (concatenated in d_out, all f32):
//   quantized_st [32768*512], ids [32768], commitment [32768*512],
//   codebook_loss [32768*512]   (commitment == codebook_loss numerically)
// ---------------------------------------------------------------------------

typedef __attribute__((ext_vector_type(16))) _Float16 v16h;
typedef __attribute__((ext_vector_type(8)))  _Float16 h8;
typedef __attribute__((ext_vector_type(8)))  float    v8f;
typedef __attribute__((ext_vector_type(4)))  float    f4;   // native vec4

#define K_CB   2048
#define D_EMB  512
#define BS_TOT 32768

#define Q_ELEMS  16777216        // 32768*512
#define IDS_OFF  16777216        // float index of ids in d_out
#define COMM_F4  4202496         // (Q_ELEMS + 32768) / 4
#define CBL_F4   8396800         // (Q_ELEMS*2 + 32768) / 4

// workspace byte offsets
#define WS_CB16   0              // 2048*512*2  = 2 MB f16 codebook
#define WS_WNORM  2097152        // 2048*4      = 8 KB ||w||^2
#define WS_IDS    2105344        // 32768*4     = 128 KB int ids

// sched_group_barrier masks
#define SGB_WMMA      0x008
#define SGB_VMEM_READ 0x020

struct BF { h8 lo, hi; };

__device__ __forceinline__ BF loadB(const _Float16* p) {
  BF r;
  r.lo = *(const h8*)p;
  r.hi = *(const h8*)(p + 16);
  return r;
}

__device__ __forceinline__ v16h mk16(h8 lo, h8 hi) {
  return __builtin_shufflevector(lo, hi, 0, 1, 2, 3, 4, 5, 6, 7,
                                 8, 9, 10, 11, 12, 13, 14, 15);
}

__device__ __forceinline__ h8 cvt8(const f4 a, const f4 b) {
  h8 r;
  r[0] = (_Float16)a.x; r[1] = (_Float16)a.y;
  r[2] = (_Float16)a.z; r[3] = (_Float16)a.w;
  r[4] = (_Float16)b.x; r[5] = (_Float16)b.y;
  r[6] = (_Float16)b.z; r[7] = (_Float16)b.w;
  return r;
}

// ---- Kernel 1: codebook -> f16 copy + per-row squared norms -----------------
// 8 waves per block, one wave per codebook row. 256 blocks * 8 = 2048 rows.
__global__ __launch_bounds__(256)
void vq_prep_codebook(const float* __restrict__ cb,
                      _Float16* __restrict__ cb16,
                      float* __restrict__ wnorm) {
  const int lane = threadIdx.x & 31;
  const int wave = threadIdx.x >> 5;
  const int k    = blockIdx.x * 8 + wave;

  const f4* row = (const f4*)(cb + (size_t)k * D_EMB) + lane * 4;
  f4 f0 = row[0], f1 = row[1], f2 = row[2], f3 = row[3];

  float s = f0.x*f0.x + f0.y*f0.y + f0.z*f0.z + f0.w*f0.w
          + f1.x*f1.x + f1.y*f1.y + f1.z*f1.z + f1.w*f1.w
          + f2.x*f2.x + f2.y*f2.y + f2.z*f2.z + f2.w*f2.w
          + f3.x*f3.x + f3.y*f3.y + f3.z*f3.z + f3.w*f3.w;

  h8* dst = (h8*)(cb16 + (size_t)k * D_EMB + lane * 16);
  dst[0] = cvt8(f0, f1);
  dst[1] = cvt8(f2, f3);

#pragma unroll
  for (int off = 16; off > 0; off >>= 1)
    s += __shfl_xor(s, off, 32);
  if (lane == 0) wnorm[k] = s;
}

// ---- Kernel 2: fused distance-matmul + running argmin -----------------------
// Block = 128 threads = 4 waves; each wave owns a 16-row block (64 rows/block,
// 512 blocks). x staged once into LDS as f16; 16 A-fragments (full D=512)
// materialized into VGPRs and held resident across all 128 column tiles.
// B fragments stream from the L2-resident f16 codebook with a depth-2
// software pipeline; sched_group_barrier enforces the "2 loads, 1 WMMA"
// issue pattern so 3 load pairs stay in flight (partial loadcnt waits).
__global__ __launch_bounds__(128)
void vq_argmin(const float* __restrict__ x,
               const _Float16* __restrict__ cb16,
               const float* __restrict__ wnorm,
               int* __restrict__ ids,
               float* __restrict__ ids_f) {
  __shared__ __attribute__((aligned(16))) _Float16 xs[64 * D_EMB];  // 64 KB

  const int tid  = threadIdx.x;
  const int lane = tid & 31;
  const int wave = tid >> 5;
  const size_t rowBase = (size_t)blockIdx.x * 64;

  // Warm the L2/L0 path to the f16 codebook while we stage x.
  __builtin_prefetch(cb16 + (size_t)(lane & 15) * D_EMB, 0, 3);

  // Stage 64 rows of x into LDS, converting f32 -> f16 on the fly
  // (x is read exactly once from HBM across the whole dispatch).
  {
    const f4* xin = (const f4*)(x + rowBase * D_EMB);
#pragma unroll 4
    for (int i = tid; i < 64 * D_EMB / 8; i += 128) {
      f4 a = xin[i * 2];
      f4 b = xin[i * 2 + 1];
      *(h8*)(xs + i * 8) = cvt8(a, b);
    }
  }
  __syncthreads();

  // A-fragment layout (ISA 16-bit A 16x32, wave32):
  //   lanes 0-15 : row M=lane,    K = {d0..d0+7, d0+16..d0+23}
  //   lanes 16-31: row M=lane-16, K = {d0+8..d0+15, d0+24..d0+31}
  const int rloc = wave * 16 + (lane & 15);
  const _Float16* ap = xs + rloc * D_EMB + ((lane < 16) ? 0 : 8);

  v16h afrag[16];
#pragma unroll
  for (int s = 0; s < 16; ++s) {
    h8 alo = *(const h8*)(ap + s * 32);
    h8 ahi = *(const h8*)(ap + s * 32 + 16);
    afrag[s] = mk16(alo, ahi);
  }

  // B-fragment source (16-bit B 32x16): lane holds column n = lane&15,
  // K = {d0..d0+7, d0+16..d0+23} (lanes<16) or {d0+8..d0+15, d0+24..d0+31}.
  const int bcol = lane & 15;
  const int bsel = (lane < 16) ? 0 : 8;
  const _Float16* bbase = cb16 + (size_t)bcol * D_EMB + bsel;

  float minv[8];
  int   mini[8];
#pragma unroll
  for (int j = 0; j < 8; ++j) { minv[j] = 3.4e38f; mini[j] = 0; }

  // Pipeline prologue: preload steps (t=0,s=0) and (t=0,s=1).
  BF pipeA = loadB(bbase);
  BF pipeB = loadB(bbase + 32);

  for (int t = 0; t < K_CB / 16; ++t) {
    const _Float16* tb = bbase + (size_t)t * (16 * D_EMB);
    const _Float16* nb = bbase + (size_t)((t + 1) & 127) * (16 * D_EMB);

    v8f acc0 = {}, acc1 = {};
    BF b0 = pipeA, b1 = pipeB, b2;
#pragma unroll
    for (int s = 0; s < 16; ++s) {
      // Issue loads for logical step s+2 (crosses into the next tile).
      const _Float16* pp = (s + 2 < 16) ? (tb + (s + 2) * 32)
                                        : (nb + (s + 2 - 16) * 32);
      b2 = loadB(pp);
      v16h b = mk16(b0.lo, b0.hi);
      if (s & 1)
        acc1 = __builtin_amdgcn_wmma_f32_16x16x32_f16(
                   false, afrag[s], false, b, (short)0, acc1, false, false);
      else
        acc0 = __builtin_amdgcn_wmma_f32_16x16x32_f16(
                   false, afrag[s], false, b, (short)0, acc0, false, false);
      b0 = b1; b1 = b2;   // rotate (SSA under full unroll)
      // Enforce "2 VMEM reads, then 1 WMMA" issue pattern per step so the
      // scheduler keeps the depth-2 pipeline (3 load pairs in flight).
      __builtin_amdgcn_sched_group_barrier(SGB_VMEM_READ, 2, 0);
      __builtin_amdgcn_sched_group_barrier(SGB_WMMA, 1, 0);
    }
    pipeA = b0; pipeB = b1;  // carry (t+1,0) and (t+1,1)

    // partial distance: ||w||^2 - 2 x.w  (||x||^2 constant per row -> dropped)
    const float wn    = wnorm[t * 16 + bcol];
    const int   colid = t * 16 + bcol;
#pragma unroll
    for (int j = 0; j < 8; ++j) {
      float sc = __builtin_fmaf(-2.0f, acc0[j] + acc1[j], wn);
      if (sc < minv[j]) { minv[j] = sc; mini[j] = colid; }
    }
  }

  // Reduce across the 16 columns held by each 16-lane half (C layout:
  // lanes 0-15 hold rows 0-7 in VGPRs 0-7, lanes 16-31 hold rows 8-15).
  // First-index tie-break to match jnp.argmin.
#pragma unroll
  for (int off = 1; off < 16; off <<= 1) {
#pragma unroll
    for (int j = 0; j < 8; ++j) {
      float ov = __shfl_xor(minv[j], off, 32);
      int   oi = __shfl_xor(mini[j], off, 32);
      if (ov < minv[j] || (ov == minv[j] && oi < mini[j])) {
        minv[j] = ov; mini[j] = oi;
      }
    }
  }

  if ((lane & 15) == 0) {
    const size_t r0 = rowBase + (size_t)wave * 16 + ((lane < 16) ? 0 : 8);
#pragma unroll
    for (int j = 0; j < 8; ++j) {
      ids[r0 + j]   = mini[j];
      ids_f[r0 + j] = (float)mini[j];
    }
  }
}

// ---- Kernel 3: gather + elementwise outputs (memory-bound, vec4) -----------
// quantized uses the ORIGINAL f32 codebook (exact values). commitment and
// codebook_loss are the same (q - x)^2, written twice. Outputs are never
// re-read -> non-temporal stores keep x/codebook resident in L2.
__global__ __launch_bounds__(256)
void vq_gather(const float* __restrict__ x,
               const float* __restrict__ cb,
               const int* __restrict__ ids,
               float* __restrict__ out) {
  const size_t i   = (size_t)blockIdx.x * blockDim.x + threadIdx.x; // f4 index
  const size_t row = i >> 7;           // / (512/4)
  const int    c4  = (int)(i & 127);
  const int    id  = ids[row];

  f4 q  = ((const f4*)cb)[(size_t)id * 128 + c4];
  f4 xv = ((const f4*)x)[i];

  f4 d  = q - xv;
  f4 d2 = d * d;

  f4* o = (f4*)out;
  __builtin_nontemporal_store(q,  o + i);            // quantized_st
  __builtin_nontemporal_store(d2, o + COMM_F4 + i);  // commitment
  __builtin_nontemporal_store(d2, o + CBL_F4 + i);   // codebook_loss
}

extern "C" void kernel_launch(void* const* d_in, const int* in_sizes, int n_in,
                              void* d_out, int out_size, void* d_ws, size_t ws_size,
                              hipStream_t stream) {
  (void)in_sizes; (void)n_in; (void)out_size; (void)ws_size;
  const float* x  = (const float*)d_in[0];
  const float* cb = (const float*)d_in[1];

  char* ws = (char*)d_ws;
  _Float16* cb16  = (_Float16*)(ws + WS_CB16);
  float*    wnorm = (float*)(ws + WS_WNORM);
  int*      ids   = (int*)(ws + WS_IDS);
  float*    out   = (float*)d_out;

  // 1) codebook -> f16 (2 MB, L2-resident thereafter) + ||w||^2
  vq_prep_codebook<<<K_CB / 8, 256, 0, stream>>>(cb, cb16, wnorm);
  // 2) fused WMMA distance + argmin (512 blocks x 64 rows)
  vq_argmin<<<BS_TOT / 64, 128, 0, stream>>>(x, cb16, wnorm, ids,
                                             out + IDS_OFF);
  // 3) gather + squared-difference outputs
  vq_gather<<<(BS_TOT * D_EMB / 4) / 256, 256, 0, stream>>>(x, cb, ids, out);
}
